// HetergatWOConcatFeat_23862838297172
// MI455X (gfx1250) — matrix operand order repaired and checked by reading, not compiled, not run on hardware
//
#include <hip/hip_runtime.h>

typedef __attribute__((ext_vector_type(16))) _Float16 v16h;
typedef __attribute__((ext_vector_type(8)))  _Float16 v8h;
typedef __attribute__((ext_vector_type(8)))  float    v8f;

typedef __attribute__((ext_vector_type(4))) unsigned int tdm_v4u;
typedef __attribute__((ext_vector_type(8))) int          tdm_v8i;
typedef __attribute__((ext_vector_type(4))) int          tdm_v4i;

#if defined(__has_builtin)
#if __has_builtin(__builtin_amdgcn_tensor_load_to_lds) && \
    __has_builtin(__builtin_amdgcn_s_wait_tensorcnt)
#define USE_TDM 1
#endif
#endif

constexpr int kNodes = 20000;   // N
constexpr int kE     = 320000;  // edges per type
constexpr int kH     = 8;       // heads
constexpr int kF0    = 128;
constexpr int kFU    = 256;
constexpr int kFO    = 64;

#ifdef USE_TDM
// Issue one TDM 2D tile load (f16 elements): rows x rowLen tile from a
// row-major tensor (row stride = strideElems) into LDS at ldsAddr.
// D# layout per CDNA5 ISA ch.8 (group0: count/lds/global/type, group1: dims).
// This toolchain's builtin takes 6 args: (g0 v4u, g1 v8i, g2 v4i, g3 v4i,
// g4 v8i, cpol i32) — extra groups zero-filled for a 2D tile.
__device__ __forceinline__ void tdm_load_tile_f16(
    const _Float16* gptr, unsigned ldsAddr, unsigned rowLen, unsigned rows,
    unsigned long long strideElems, unsigned tensorW, unsigned tensorH)
{
  unsigned long long ga = (unsigned long long)(uintptr_t)gptr;
  tdm_v4u g0;
  g0[0] = 1u;                                          // count=1, user desc
  g0[1] = ldsAddr;                                     // LDS byte address
  g0[2] = (unsigned)(ga & 0xFFFFFFFFu);                // global_addr[31:0]
  g0[3] = (unsigned)((ga >> 32) & 0x01FFFFFFu)         // global_addr[56:32]
        | 0x80000000u;                                 // type=2 ("image")
  tdm_v8i g1;
  g1[0] = (int)(1u << 16);                             // data_size=1 (2 bytes)
  g1[1] = (int)((tensorW & 0xFFFFu) << 16);            // tensor_dim0 lo16
  g1[2] = (int)((tensorW >> 16) | ((tensorH & 0xFFFFu) << 16));
  g1[3] = (int)((tensorH >> 16) | (rowLen << 16));     // tile_dim0
  g1[4] = (int)rows;                                   // tile_dim1 (tile_dim2=0)
  g1[5] = (int)(unsigned)(strideElems & 0xFFFFFFFFull);// dim0_stride lo32
  g1[6] = (int)(unsigned)((strideElems >> 32) & 0xFFFFull);
  g1[7] = 0;
  tdm_v4i z4 = {0, 0, 0, 0};
  tdm_v8i z8 = {0, 0, 0, 0, 0, 0, 0, 0};
  __builtin_amdgcn_tensor_load_to_lds(g0, g1, z4, z4, z8, 0);
}
#endif

// ---------------------------------------------------------------------------
// Batched WMMA GEMM: C[b] = A (MxK, f16) * B[b] (KxN, f32), f32 out.
// Block = 256 threads (8 waves) computes a 128x64 C tile.
// Per K-step (BK=32): A tile DMA'd to LDS by the Tensor Data Mover (wave 0),
// B tile staged f32->f16 transposed; each wave does 4 x v_wmma_f32_16x16x32_f16.
// Requirements: K%32==0, N%64==0, lda/ldb multiples of 4. M guarded on store.
// ---------------------------------------------------------------------------
__global__ __launch_bounds__(256) void wmma_gemm_batched(
    const _Float16* __restrict__ A, int lda,
    const float* __restrict__ B, int ldb, long long strideB,
    float* __restrict__ C, int ldc, long long strideC,
    int M, int K)
{
  __shared__ __align__(32) _Float16 As[128 * 32];  // [row][k]
  __shared__ __align__(32) _Float16 Bs[64 * 32];   // [col][k]  (transposed)

  const int tid  = threadIdx.x;
  const int lane = tid & 31;
  const int wave = tid >> 5;
  const int wm   = wave & 3;     // 4 waves along M
  const int wn   = wave >> 2;    // 2 waves along N
  const int half = lane >> 4;    // 0: lanes 0-15, 1: lanes 16-31
  const int l16  = lane & 15;

  const long long bz = blockIdx.z;
  const float* Bb = B + bz * strideB;
  float*       Cb = C + bz * strideC;

  const int row0 = blockIdx.x * 128;
  const int col0 = blockIdx.y * 64;

  // B staging coords
  const int bcol  = (tid & 15) * 4;  // 0..60
  const int brow0 = tid >> 4;        // 0..15
#ifndef USE_TDM
  const int arow  = tid >> 1;              // 0..127
  const int acol0 = (tid & 1) * 16;        // 0 or 16
#endif

  v8f acc[2][2] = {{{}, {}}, {{}, {}}};

  for (int k0 = 0; k0 < K; k0 += 32) {
    // ---- stage A tile (128x32 f16) ----
#ifdef USE_TDM
    if (wave == 0)
      tdm_load_tile_f16(A + (long long)row0 * lda + k0,
                        (unsigned)(uintptr_t)&As[0], 32u, 128u,
                        (unsigned long long)lda, (unsigned)K, (unsigned)M);
#else
    {
      long long gr = row0 + arow;
      if (gr > M - 1) gr = M - 1;
      const uint4* src = (const uint4*)(A + gr * lda + k0 + acol0);
      uint4 w0 = src[0], w1 = src[1];
      uint4* dst = (uint4*)&As[arow * 32 + acol0];
      dst[0] = w0; dst[1] = w1;
    }
#endif
    // ---- stage B tile (32x64 f32 -> f16, transposed to [col][k]) ----
    {
      float4 bv[2];
#pragma unroll
      for (int i = 0; i < 2; ++i)
        bv[i] = *(const float4*)(Bb + (long long)(k0 + brow0 + 16 * i) * ldb +
                                 col0 + bcol);
#pragma unroll
      for (int i = 0; i < 2; ++i) {
        int k = brow0 + 16 * i;
        Bs[(bcol + 0) * 32 + k] = (_Float16)bv[i].x;
        Bs[(bcol + 1) * 32 + k] = (_Float16)bv[i].y;
        Bs[(bcol + 2) * 32 + k] = (_Float16)bv[i].z;
        Bs[(bcol + 3) * 32 + k] = (_Float16)bv[i].w;
      }
    }
    // prefetch next B tile (global_prefetch_b8)
    if (k0 + 32 < K)
      __builtin_prefetch(Bb + (long long)(k0 + 32 + brow0) * ldb + col0 + bcol,
                         0, 1);
#ifdef USE_TDM
    if (wave == 0) __builtin_amdgcn_s_wait_tensorcnt(0);
#endif
    __syncthreads();

    // ---- fragments from LDS ----
    v16h a[2], b[2];
#pragma unroll
    for (int mi = 0; mi < 2; ++mi) {
      const _Float16* ap = &As[(wm * 32 + mi * 16 + l16) * 32 + half * 8];
      v8h lo = *(const v8h*)ap;          // K = k0 + half*8 + 0..7
      v8h hi = *(const v8h*)(ap + 16);   // K = k0 + 16 + half*8 + 0..7
#pragma unroll
      for (int i = 0; i < 8; ++i) { a[mi][i] = lo[i]; a[mi][8 + i] = hi[i]; }
    }
#pragma unroll
    for (int ni = 0; ni < 2; ++ni) {
      const _Float16* bp = &Bs[(wn * 32 + ni * 16 + l16) * 32 + half * 8];
      v8h lo = *(const v8h*)bp;
      v8h hi = *(const v8h*)(bp + 16);
#pragma unroll
      for (int i = 0; i < 8; ++i) { b[ni][i] = lo[i]; b[ni][8 + i] = hi[i]; }
    }

    // ---- 4 WMMAs ----
#pragma unroll
    for (int mi = 0; mi < 2; ++mi)
#pragma unroll
      for (int ni = 0; ni < 2; ++ni)
        acc[mi][ni] = __builtin_amdgcn_wmma_f32_16x16x32_f16(
            false, a[mi], false, b[ni], (short)0, acc[mi][ni], false, false);

    __syncthreads();
  }

  // ---- store C (guarded rows) ----
#pragma unroll
  for (int mi = 0; mi < 2; ++mi)
#pragma unroll
    for (int ni = 0; ni < 2; ++ni)
#pragma unroll
      for (int r = 0; r < 8; ++r) {
        long long row = row0 + wm * 32 + mi * 16 + r + half * 8;
        if (row < M)
          Cb[row * ldc + col0 + wn * 32 + ni * 16 + l16] = acc[mi][ni][r];
      }
}

// ---------------------------------------------------------------------------
__global__ void cvt_f32_f16(const float* __restrict__ in,
                            _Float16* __restrict__ out, long long n)
{
  long long i = blockIdx.x * (long long)blockDim.x + threadIdx.x;
  if (i < n) out[i] = (_Float16)in[i];
}

// s[n][h] = <hp[h][n][:], asrc[h][:]> ; t[n][h] = <hp[h][n][:], atrg[h][:]>
__global__ void attn_scores(const float* __restrict__ hp,   // [H][N][64]
                            const float* __restrict__ asrc, // [H][64]
                            const float* __restrict__ atrg, // [H][64]
                            float* __restrict__ s, float* __restrict__ t, int n)
{
  int i = blockIdx.x * blockDim.x + threadIdx.x;
  if (i >= n * kH) return;
  int node = i / kH, h = i % kH;
  const float* v  = hp + ((long long)h * n + node) * kFO;
  const float* as = asrc + h * kFO;
  const float* at = atrg + h * kFO;
  float ss = 0.f, tt = 0.f;
#pragma unroll 4
  for (int o = 0; o < kFO; ++o) { float x = v[o]; ss += x * as[o]; tt += x * at[o]; }
  s[(long long)node * kH + h] = ss;
  t[(long long)node * kH + h] = tt;
}

__global__ void zero_f32(float* __restrict__ p, long long n)
{
  long long i = blockIdx.x * (long long)blockDim.x + threadIdx.x;
  if (i < n) p[i] = 0.f;
}

__global__ void zero_u32(unsigned* p) { *p = 0u; }

__device__ inline unsigned flipf(float f) {
  unsigned u = __float_as_uint(f);
  return (u >> 31) ? ~u : (u | 0x80000000u);
}
__device__ inline float unflipf(unsigned u) {
  return __uint_as_float((u >> 31) ? (u & 0x7FFFFFFFu) : ~u);
}

// Pass 1: e = leaky_relu(s[src]+t[trg]); global max into *gmax (flipped uint)
__global__ void edge_pass1(const int* __restrict__ src, const int* __restrict__ trg,
                           const float* __restrict__ s, const float* __restrict__ t,
                           float* __restrict__ e, unsigned* __restrict__ gmax, int nE)
{
  __shared__ unsigned smax;
  if (threadIdx.x == 0) smax = 0u;
  __syncthreads();
  int i = blockIdx.x * blockDim.x + threadIdx.x;
  float lmax = -3.4e38f;
  if (i < nE) {
    int a = src[i], b = trg[i];
#pragma unroll
    for (int h = 0; h < kH; ++h) {
      float v = s[(long long)a * kH + h] + t[(long long)b * kH + h];
      v = v > 0.f ? v : 0.2f * v;
      e[(long long)i * kH + h] = v;
      lmax = fmaxf(lmax, v);
    }
  }
  atomicMax(&smax, flipf(lmax));
  __syncthreads();
  if (threadIdx.x == 0) atomicMax(gmax, smax);
}

// Pass 2: ex = exp(e - max); denom[trg][h] += ex (segment sum)
__global__ void edge_pass2(const int* __restrict__ trg, float* __restrict__ e,
                           const unsigned* __restrict__ gmax,
                           float* __restrict__ denom, int nE)
{
  int i = blockIdx.x * blockDim.x + threadIdx.x;
  if (i >= nE) return;
  float m = unflipf(*gmax);
  int b = trg[i];
#pragma unroll
  for (int h = 0; h < kH; ++h) {
    float ex = expf(e[(long long)i * kH + h] - m);
    e[(long long)i * kH + h] = ex;
    unsafeAtomicAdd(&denom[(long long)b * kH + h], ex);
  }
}

// Pass 3: acc[trg][h][:] += hp[h][src][:] * (ex / (denom[trg][h] + 1e-16))
// One wave per edge; lane handles columns lane and lane+32.
__global__ __launch_bounds__(256) void edge_pass3(
    const int* __restrict__ src, const int* __restrict__ trg,
    const float* __restrict__ ex, const float* __restrict__ denom,
    const float* __restrict__ hp,   // [H][N][64]
    float* __restrict__ acc,        // [N][H][64]
    int n, int nE)
{
  int lane = threadIdx.x & 31;
  int ei = blockIdx.x * 8 + (threadIdx.x >> 5);
  if (ei >= nE) return;
  int a = src[ei], b = trg[ei];
#pragma unroll
  for (int h = 0; h < kH; ++h) {
    float alpha = ex[(long long)ei * kH + h] /
                  (denom[(long long)b * kH + h] + 1e-16f);
    const float* v = hp + ((long long)h * n + a) * kFO;
    float* o = acc + ((long long)b * kH + h) * kFO;
    unsafeAtomicAdd(&o[lane],      v[lane]      * alpha);
    unsafeAtomicAdd(&o[lane + 32], v[lane + 32] * alpha);
  }
}

// elu; writes f16 (consumed only as next GEMM's A operand)
__global__ void elu_to_f16(const float* __restrict__ in,
                           _Float16* __restrict__ out, long long n)
{
  long long i = blockIdx.x * (long long)blockDim.x + threadIdx.x;
  if (i >= n) return;
  float v = in[i];
  out[i] = (_Float16)(v > 0.f ? v : (expf(v) - 1.f));
}

// mean over heads: ta[n][typ][o] = mean_h acc[n][h][o]
__global__ void head_mean(const float* __restrict__ acc, float* __restrict__ ta,
                          int typ, int n)
{
  int i = blockIdx.x * blockDim.x + threadIdx.x;
  if (i >= n * kFO) return;
  int node = i >> 6, o = i & 63;
  const float* p = acc + (long long)node * kH * kFO + o;
  float s = 0.f;
#pragma unroll
  for (int h = 0; h < kH; ++h) s += p[h * kFO];
  ta[((long long)node * 2 + typ) * kFO + o] = s * 0.125f;
}

// favec[o] = trans1[0][:] @ aw1[:,o]
__global__ void feat_vec(const float* __restrict__ trans1,
                         const float* __restrict__ aw1, float* __restrict__ favec)
{
  int o = threadIdx.x;
  float s = 0.f;
  for (int f = 0; f < kFU; ++f) s += trans1[f] * aw1[f * kFO + o];
  favec[o] = s;
}

// Semantic attention + classifier + log_softmax. One wave per node.
__global__ __launch_bounds__(256) void final_fuse(
    const float* __restrict__ ta,    // [N][2][64]
    const float* __restrict__ favec, // [64]
    const float* __restrict__ aw2,   // [64][64]
    const float* __restrict__ am,    // [64]
    const float* __restrict__ fcw,   // [192][2]
    const float* __restrict__ fcb,   // [2]
    float* __restrict__ out, int n)
{
  int lane = threadIdx.x & 31;
  int node = blockIdx.x * 8 + (threadIdx.x >> 5);
  if (node >= n) return;
  const float* t0 = ta + (long long)node * 2 * kFO;
  const float* t1 = t0 + kFO;

  float qa[2];
#pragma unroll
  for (int r = 0; r < 2; ++r) {
    const float* tr = r ? t1 : t0;
    float part = 0.f;
#pragma unroll
    for (int oo = 0; oo < 2; ++oo) {
      int o = lane + oo * 32;
      float a = favec[o];
      for (int k = 0; k < kFO; ++k) a += tr[k] * aw2[k * kFO + o];
      part += tanhf(a) * am[o];
    }
    for (int off = 16; off; off >>= 1) part += __shfl_xor(part, off, 32);
    qa[r] = part;
  }
  float m  = fmaxf(qa[0], qa[1]);
  float e0 = expf(qa[0] - m), e1 = expf(qa[1] - m);
  float inv = 1.f / (e0 + e1);
  float b0 = e0 * inv, b1 = e1 * inv;

  float l0 = 0.f, l1 = 0.f;
#pragma unroll
  for (int oo = 0; oo < 2; ++oo) {
    int o = lane + oo * 32;
    float v0 = t0[o], v1 = t1[o];
    float f  = b0 * v0 + b1 * v1;
    l0 += v0 * fcw[o * 2] + v1 * fcw[(64 + o) * 2] + f * fcw[(128 + o) * 2];
    l1 += v0 * fcw[o * 2 + 1] + v1 * fcw[(64 + o) * 2 + 1] + f * fcw[(128 + o) * 2 + 1];
  }
  for (int off = 16; off; off >>= 1) {
    l0 += __shfl_xor(l0, off, 32);
    l1 += __shfl_xor(l1, off, 32);
  }
  if (lane == 0) {
    l0 += fcb[0]; l1 += fcb[1];
    float mm  = fmaxf(l0, l1);
    float lse = mm + logf(expf(l0 - mm) + expf(l1 - mm));
    out[(long long)node * 2 + 0] = l0 - lse;
    out[(long long)node * 2 + 1] = l1 - lse;
  }
}

// ---------------------------------------------------------------------------
extern "C" void kernel_launch(void* const* d_in, const int* in_sizes, int n_in,
                              void* d_out, int out_size, void* d_ws, size_t ws_size,
                              hipStream_t stream)
{
  (void)in_sizes; (void)n_in; (void)out_size; (void)ws_size;
  const float* hemb[2] = { (const float*)d_in[0], (const float*)d_in[1] };
  const float* W[2]    = { (const float*)d_in[2], (const float*)d_in[3] };
  const float* aw1 = (const float*)d_in[16];
  const float* aw2 = (const float*)d_in[17];
  const float* am  = (const float*)d_in[18];
  const float* fcw = (const float*)d_in[19];
  const float* fcb = (const float*)d_in[20];
  float* out = (float*)d_out;

  // workspace layout (float-sized slots; f16 buffers use half a slot/elem)
  float* ws = (float*)d_ws;
  size_t off = 0;
  float* trans[2];
  trans[0] = ws + off; off += (size_t)kNodes * kFU;
  trans[1] = ws + off; off += (size_t)kNodes * kFU;
  float* hp    = ws + off; off += (size_t)kH * kNodes * kFO;
  float* acc   = ws + off; off += (size_t)kNodes * kH * kFO;
  float* sb    = ws + off; off += (size_t)kNodes * kH;
  float* tb    = ws + off; off += (size_t)kNodes * kH;
  float* eb    = ws + off; off += (size_t)kE * kH;
  float* denom = ws + off; off += (size_t)kNodes * kH;
  float* tavg  = ws + off; off += (size_t)kNodes * 2 * kFO;
  float* favec = ws + off; off += 64;
  unsigned* gmax = (unsigned*)(ws + off); off += 1;
  _Float16* hembh[2];
  hembh[0] = (_Float16*)(ws + off); off += (size_t)kNodes * kF0 / 2;
  hembh[1] = (_Float16*)(ws + off); off += (size_t)kNodes * kF0 / 2;
  _Float16* transh = (_Float16*)(ws + off); off += (size_t)kNodes * kFU / 2;
  _Float16* xh     = (_Float16*)(ws + off); off += (size_t)kNodes * kH * kFO / 2;

  const long long accN = (long long)kNodes * kH * kFO;
  const int mBlocks = (kNodes + 127) / 128;   // 157

  // trans_t = hemb_t @ W_t : M=20000 N=256 K=128 (A pre-converted to f16)
  for (int t = 0; t < 2; ++t) {
    const long long nH = (long long)kNodes * kF0;
    cvt_f32_f16<<<dim3((unsigned)((nH + 255) / 256)), 256, 0, stream>>>(
        hemb[t], hembh[t], nH);
    wmma_gemm_batched<<<dim3(mBlocks, kFU / 64, 1), 256, 0, stream>>>(
        hembh[t], kF0, W[t], kFU, 0, trans[t], kFU, 0, kNodes, kF0);
  }

  for (int t = 0; t < 2; ++t) {
    const int* edge = (const int*)d_in[21 + t];
    const int* src = edge;
    const int* trg = edge + kE;

    const long long nT = (long long)kNodes * kFU;
    cvt_f32_f16<<<dim3((unsigned)((nT + 255) / 256)), 256, 0, stream>>>(
        trans[t], transh, nT);

    for (int l = 0; l < 2; ++l) {
      const int fin = (l == 0) ? kFU : kH * kFO;
      const _Float16* Ain = (l == 0) ? transh : xh;
      const int wi = 4 + t * 6 + l * 3;
      const float* gw   = (const float*)d_in[wi];
      const float* gas  = (const float*)d_in[wi + 1];
      const float* gat  = (const float*)d_in[wi + 2];

      // hp[h] = Ain @ gw[h] : M=20000 N=64 K=fin, batch=8 heads
      wmma_gemm_batched<<<dim3(mBlocks, 1, 8), 256, 0, stream>>>(
          Ain, fin, gw, kFO, (long long)fin * kFO,
          hp, kFO, (long long)kNodes * kFO, kNodes, fin);

      attn_scores<<<dim3((kNodes * kH + 255) / 256), 256, 0, stream>>>(
          hp, gas, gat, sb, tb, kNodes);

      zero_f32<<<dim3((unsigned)((accN + 255) / 256)), 256, 0, stream>>>(acc, accN);
      zero_f32<<<dim3((kNodes * kH + 255) / 256), 256, 0, stream>>>(denom,
                                                     (long long)kNodes * kH);
      zero_u32<<<1, 1, 0, stream>>>(gmax);

      edge_pass1<<<dim3((kE + 255) / 256), 256, 0, stream>>>(src, trg, sb, tb,
                                                             eb, gmax, kE);
      edge_pass2<<<dim3((kE + 255) / 256), 256, 0, stream>>>(trg, eb, gmax,
                                                             denom, kE);
      edge_pass3<<<dim3((kE + 7) / 8), 256, 0, stream>>>(src, trg, eb, denom,
                                                         hp, acc, kNodes, kE);
      if (l == 0)
        elu_to_f16<<<dim3((unsigned)((accN + 255) / 256)), 256, 0, stream>>>(
            acc, xh, accN);
      else
        head_mean<<<dim3((kNodes * kFO + 255) / 256), 256, 0, stream>>>(acc, tavg,
                                                                        t, kNodes);
    }
  }

  feat_vec<<<1, 64, 0, stream>>>(trans[1], aw1, favec);
  final_fuse<<<dim3((kNodes + 7) / 8), 256, 0, stream>>>(tavg, favec, aw2, am,
                                                         fcw, fcb, out, kNodes);
}
